// Generator_19292993094051
// MI455X (gfx1250) — matrix operand order repaired and checked by reading, not compiled, not run on hardware
//
#include <hip/hip_runtime.h>
#include <hip/hip_bf16.h>

typedef __attribute__((ext_vector_type(8)))  __bf16 bf16x8;
typedef __attribute__((ext_vector_type(16))) __bf16 bf16x16;
typedef __attribute__((ext_vector_type(8)))  float  f32x8;

#define LEAKC 0.2f
#define BATCH 16

__device__ __forceinline__ float leakyf(float v) { return v >= 0.f ? v : LEAKC * v; }

// Hash-based approx N(0,1): jax PRNG is not reproducible on-device; noise_w = 0.01.
__device__ __forceinline__ float noise_val(int li, int b, int h, int w) {
  unsigned x = (unsigned)(li + 1) * 1315423911u;
  x ^= (unsigned)b * 2654435761u;
  x ^= (unsigned)h * 2246822519u;
  x ^= (unsigned)w * 3266489917u;
  x ^= x >> 16; x *= 0x7feb352du; x ^= x >> 15; x *= 0x846ca68bu; x ^= x >> 16;
  unsigned y = x * 0x9E3779B9u + 0x85EBCA6Bu;
  y ^= y >> 16; y *= 0x21f0aaadu; y ^= y >> 15; y *= 0x735a2d97u; y ^= y >> 15;
  float u1 = (float)(x >> 8) * (1.0f / 16777216.0f) + 1e-7f;
  float u2 = (float)(y >> 8) * (1.0f / 16777216.0f);
  return sqrtf(-2.f * logf(u1)) * cosf(6.28318530718f * u2);
}

// ---------------- elementwise helpers ----------------

__global__ void k_pack_bf16(const float* __restrict__ s, __bf16* __restrict__ d, int n) {
  int i = blockIdx.x * blockDim.x + threadIdx.x;
  if (i < n) d[i] = (__bf16)s[i];
}

__global__ void k_copyf(const float* __restrict__ s, float* __restrict__ d, int n) {
  int i = blockIdx.x * blockDim.x + threadIdx.x;
  if (i < n) d[i] = s[i];
}

// learned_input (1,512,4,4) NCHW -> x0 NHWC (16,4,4,512), broadcast over batch
__global__ void k_init(const float* __restrict__ li, float* __restrict__ x0) {
  int i = blockIdx.x * blockDim.x + threadIdx.x;
  int n = BATCH * 4 * 4 * 512;
  if (i >= n) return;
  int c = i % 512; int t = i / 512;
  int w = t % 4; t /= 4;
  int h = t % 4;
  x0[i] = li[(c * 4 + h) * 4 + w];
}

// style[b,c] = w_lat[b,:] @ style_w[c,:] + style_b[c] + 1  (zero for padded c)
__global__ void k_style(const float* __restrict__ wlat, const float* __restrict__ sw,
                        const float* __restrict__ sb, float* __restrict__ style,
                        int Cin, int Cpad) {
  int i = blockIdx.x * blockDim.x + threadIdx.x;
  if (i >= BATCH * Cpad) return;
  int b = i / Cpad, c = i % Cpad;
  float s = 0.f;
  if (c < Cin) {
    const float* wr = wlat + b * 512;
    const float* sr = sw + (size_t)c * 512;
    float acc = 0.f;
    for (int k = 0; k < 512; ++k) acc += wr[k] * sr[k];
    s = acc + sb[c] + 1.f;
  }
  style[i] = s;
}

// conv_w (O,I,3,3) f32 -> wpk[tap][O][Ipad] bf16 ; wsq[O][Ipad] = sum_tap w^2
__global__ void k_prepack(const float* __restrict__ cw, __bf16* __restrict__ wpk,
                          float* __restrict__ wsq, int Cout, int Cin, int Cpad) {
  int i = blockIdx.x * blockDim.x + threadIdx.x;
  if (i >= Cout * Cpad) return;
  int o = i / Cpad, ci = i % Cpad;
  float sq = 0.f;
  for (int t = 0; t < 9; ++t) {
    float v = 0.f;
    if (ci < Cin) v = cw[(size_t)(o * Cin + ci) * 9 + t];
    wpk[((size_t)t * Cout + o) * Cpad + ci] = (__bf16)v;
    sq += v * v;
  }
  wsq[i] = sq;
}

// demod[b,o] = 1 / max(sqrt(sum_i style^2 * wsq), 1e-12)
__global__ void k_demod(const float* __restrict__ style, const float* __restrict__ wsq,
                        float* __restrict__ demod, int Cout, int Cpad) {
  int i = blockIdx.x * blockDim.x + threadIdx.x;
  if (i >= BATCH * Cout) return;
  int b = i / Cout, o = i % Cout;
  const float* sp = style + b * Cpad;
  const float* qp = wsq + (size_t)o * Cpad;
  float acc = 0.f;
  for (int c = 0; c < Cpad; ++c) acc += sp[c] * sp[c] * qp[c];
  demod[i] = 1.f / fmaxf(sqrtf(acc), 1e-12f);
}

// xs NHWC(bf16, Cpad) = (optional up2 of) x NHWC(f32, Cx) * style
__global__ void k_modulate(const float* __restrict__ x, const float* __restrict__ style,
                           __bf16* __restrict__ xs, int Hin, int Cx,
                           int Hout, int Cpad, int up) {
  int i = blockIdx.x * blockDim.x + threadIdx.x;
  int n = BATCH * Hout * Hout * Cpad;
  if (i >= n) return;
  int c = i % Cpad; int t = i / Cpad;
  int w = t % Hout; t /= Hout;
  int h = t % Hout;
  int b = t / Hout;
  float v = 0.f;
  if (c < Cx) {
    int hs = up ? (h >> 1) : h;
    int ws = up ? (w >> 1) : w;
    v = x[(((size_t)b * Hin + hs) * Hin + ws) * Cx + c] * style[b * Cpad + c];
  }
  xs[i] = (__bf16)v;
}

// depthwise [1,3,3,1]^2 /64 blur, pad (1,2); up=1: input is half-res (nearest up2 fused)
__global__ void k_blur(const float* __restrict__ in, const float* __restrict__ add,
                       float* __restrict__ out, int H, int C, int up) {
  int i = blockIdx.x * blockDim.x + threadIdx.x;
  int n = BATCH * H * H * C;
  if (i >= n) return;
  int c = i % C; int t = i / C;
  int w = t % H; t /= H;
  int h = t % H;
  int b = t / H;
  const float kk[4] = {1.f, 3.f, 3.f, 1.f};
  int Hi = up ? (H >> 1) : H;
  float acc = 0.f;
  for (int dy = 0; dy < 4; ++dy) {
    int hh = h + dy - 1;
    if (hh < 0 || hh >= H) continue;
    int hs = up ? (hh >> 1) : hh;
    for (int dx = 0; dx < 4; ++dx) {
      int ww = w + dx - 1;
      if (ww < 0 || ww >= H) continue;
      int ws = up ? (ww >> 1) : ww;
      acc += kk[dy] * kk[dx] * in[(((size_t)b * Hi + hs) * Hi + ws) * C + c];
    }
  }
  acc *= (1.f / 64.f);
  if (add) acc += add[i];
  out[i] = acc;
}

// x = leaky(y + bias + noise*nw)
__global__ void k_finish(const float* __restrict__ yin, const float* __restrict__ bias,
                         const float* __restrict__ nwp, float* __restrict__ xout,
                         int H, int C, int li) {
  int i = blockIdx.x * blockDim.x + threadIdx.x;
  int n = BATCH * H * H * C;
  if (i >= n) return;
  int c = i % C; int t = i / C;
  int w = t % H; t /= H;
  int h = t % H;
  int b = t / H;
  float v = yin[i] + bias[c] + noise_val(li, b, h, w) * nwp[0];
  xout[i] = leakyf(v);
}

// to-RGB: 1x1 conv Cout=3, modulation fused, no demod, no act
__global__ void k_rgb(const float* __restrict__ x, const float* __restrict__ style,
                      const float* __restrict__ cw, const float* __restrict__ cb,
                      const float* __restrict__ nwp, float* __restrict__ rout,
                      int H, int Cin, int li) {
  int i = blockIdx.x * blockDim.x + threadIdx.x;
  int n = BATCH * H * H;
  if (i >= n) return;
  int w = i % H;
  int h = (i / H) % H;
  int b = i / (H * H);
  const float* xp = x + (size_t)i * Cin;
  const float* sp = style + b * Cin;
  float a0 = 0.f, a1 = 0.f, a2 = 0.f;
  for (int c = 0; c < Cin; ++c) {
    float xv = xp[c] * sp[c];
    a0 += xv * cw[c];
    a1 += xv * cw[Cin + c];
    a2 += xv * cw[2 * Cin + c];
  }
  float nv = noise_val(li, b, h, w) * nwp[0];
  rout[(size_t)i * 3 + 0] = a0 + cb[0] + nv;
  rout[(size_t)i * 3 + 1] = a1 + cb[1] + nv;
  rout[(size_t)i * 3 + 2] = a2 + cb[2] + nv;
}

// y NHWC (16,256,256,3) -> d_out NCHW (16,3,256,256)
__global__ void k_output(const float* __restrict__ y, float* __restrict__ out) {
  int i = blockIdx.x * blockDim.x + threadIdx.x;
  int n = BATCH * 3 * 256 * 256;
  if (i >= n) return;
  int w = i % 256; int t = i / 256;
  int h = t % 256; t /= 256;
  int c = t % 3;
  int b = t / 3;
  out[i] = y[(((size_t)b * 256 + h) * 256 + w) * 3 + c];
}

// ---------------- WMMA kernels ----------------

// mapping MLP layer: out[16,512] = leaky(act[16,512] @ W^T + b), bf16 WMMA f32 acc
__global__ __launch_bounds__(32)
void k_map_layer(const __bf16* __restrict__ act_in, const __bf16* __restrict__ Wbf,
                 const float* __restrict__ bias, __bf16* __restrict__ act_out,
                 float* __restrict__ wlat_out, int is_last) {
  const int lane = threadIdx.x & 31;
  const int m = lane & 15;
  const int half = lane >> 4;
  const int n0 = blockIdx.x * 64;
  f32x8 acc[4] = {};
  for (int kc = 0; kc < 512; kc += 32) {
    const __bf16* ap = act_in + m * 512 + kc + half * 8;
    bf16x8 alo = *(const bf16x8*)(ap);
    bf16x8 ahi = *(const bf16x8*)(ap + 16);
    bf16x16 A = __builtin_shufflevector(alo, ahi, 0, 1, 2, 3, 4, 5, 6, 7,
                                        8, 9, 10, 11, 12, 13, 14, 15);
#pragma unroll
    for (int f = 0; f < 4; ++f) {
      const __bf16* bp = Wbf + (size_t)(n0 + f * 16 + m) * 512 + kc + half * 8;
      bf16x8 blo = *(const bf16x8*)(bp);
      bf16x8 bhi = *(const bf16x8*)(bp + 16);
      bf16x16 B = __builtin_shufflevector(blo, bhi, 0, 1, 2, 3, 4, 5, 6, 7,
                                          8, 9, 10, 11, 12, 13, 14, 15);
      acc[f] = __builtin_amdgcn_wmma_f32_16x16x32_bf16(false, A, false, B,
                                                       (short)0, acc[f], false, false);
    }
  }
#pragma unroll
  for (int f = 0; f < 4; ++f) {
    int nn = n0 + f * 16 + m;
#pragma unroll
    for (int r = 0; r < 8; ++r) {
      int mm = half * 8 + r;
      float v = leakyf(acc[f][r] + bias[nn]);
      act_out[mm * 512 + nn] = (__bf16)v;
      if (is_last) wlat_out[mm * 512 + nn] = v;
    }
  }
}

// Implicit-GEMM 3x3 modulated conv, NF = compile-time N-fragment count per wave.
// Weight tiles (shared by all 8 waves of the block) are staged into LDS with
// double-buffered GLOBAL_LOAD_ASYNC_TO_LDS_B128 (ASYNCcnt), consumed via ds_load_b128.
// Activation fragments are software-pipelined one K-iteration ahead.
// xs: NHWC bf16 (16,H,H,Cpad), wpk: [tap][Cout][Cpad] bf16, out: NHWC f32.
template <int NF>
__global__ __launch_bounds__(256)
void k_conv3(const __bf16* __restrict__ xs, const __bf16* __restrict__ wpk,
             const float* __restrict__ demod, float* __restrict__ yraw,
             int H, int Cpad, int Cout) {
  __shared__ __align__(16) __bf16 smw[2][2048];  // 2 x 4KB weight staging buffers

  const int tid = threadIdx.x;
  const int lane = tid & 31;
  const int wid = tid >> 5;
  const int tile = blockIdx.x * 8 + wid;
  const int m = lane & 15;
  const int half = lane >> 4;
  const int HW = H * H;
  const int p = tile * 16 + m;
  const int b = p / HW;
  const int rem = p - b * HW;
  const int h = rem / H;
  const int w = rem - h * H;
  const int n0 = blockIdx.y * (NF * 16);
  const int ka = half * 8;   // A-frag K sub-offset (lanes 16-31: K=8..15,24..31)
  const int kb = half * 8;   // B-frag K sub-offset (lanes 16-31: K=8..15,24..31)
  const int nkc = Cpad >> 5;
  const int niter = 9 * nkc;

  // this thread's 16B slice of the (tap, kc) weight chunk: linear elem = tid*8
  const int sn = tid >> 2;          // cout within chunk [0, 64)
  const int sk = (tid & 3) * 8;     // k within chunk
  const bool sact = sn < NF * 16;
  const unsigned ldsA = (unsigned)(size_t)(&smw[0][0]);
  const unsigned ldsB = (unsigned)(size_t)(&smw[1][0]);

  auto stage = [&](int it) {
    int t = it / nkc;
    int kc = (it - t * nkc) << 5;
    if (sact) {
      const __bf16* g = wpk + (size_t)(t * Cout + n0 + sn) * Cpad + kc + sk;
      unsigned lds = ((it & 1) ? ldsB : ldsA) + (unsigned)(tid * 16);
      asm volatile("global_load_async_to_lds_b128 %0, %1, off"
                   :: "v"(lds), "v"((unsigned long long)(size_t)g)
                   : "memory");
    }
  };

  auto loadA = [&](int it, bf16x8& lo, bf16x8& hi) {
    int t = it / nkc;
    int kc = (it - t * nkc) << 5;
    int ty = t / 3, tx = t - ty * 3;
    int hh = h + ty - 1, ww = w + tx - 1;
    bool valid = (hh >= 0) && (hh < H) && (ww >= 0) && (ww < H);
    lo = bf16x8{}; hi = bf16x8{};
    if (valid) {
      const __bf16* ap = xs + ((size_t)(b * H + hh) * H + ww) * Cpad + kc + ka;
      lo = *(const bf16x8*)(ap);
      hi = *(const bf16x8*)(ap + 16);
    }
  };

  f32x8 acc[NF] = {};

  // prologue: stage first weight chunk, preload first A fragment
  stage(0);
  bf16x8 alo, ahi;
  loadA(0, alo, ahi);
  asm volatile("s_wait_asynccnt 0x0" ::: "memory");
  __syncthreads();

  for (int it = 0; it < niter; ++it) {
    const bool more = (it + 1 < niter);
    if (more) stage(it + 1);               // async -> other LDS buffer (overlaps WMMA)
    bf16x8 nlo = bf16x8{}, nhi = bf16x8{};
    if (more) loadA(it + 1, nlo, nhi);     // pipelined activation fragment

    const __bf16* sb = (it & 1) ? &smw[1][0] : &smw[0][0];
    bf16x16 A = __builtin_shufflevector(alo, ahi, 0, 1, 2, 3, 4, 5, 6, 7,
                                        8, 9, 10, 11, 12, 13, 14, 15);
#pragma unroll
    for (int f = 0; f < NF; ++f) {
      const __bf16* bp = sb + (f * 16 + m) * 32 + kb;
      bf16x8 blo = *(const bf16x8*)(bp);       // ds_load_b128
      bf16x8 bhi = *(const bf16x8*)(bp + 16);  // ds_load_b128
      bf16x16 B = __builtin_shufflevector(blo, bhi, 0, 1, 2, 3, 4, 5, 6, 7,
                                          8, 9, 10, 11, 12, 13, 14, 15);
      acc[f] = __builtin_amdgcn_wmma_f32_16x16x32_bf16(false, A, false, B,
                                                       (short)0, acc[f], false, false);
    }
    alo = nlo; ahi = nhi;

    if (more) {
      asm volatile("s_wait_asynccnt 0x0" ::: "memory");  // own slice landed
      __syncthreads();                                   // all waves' slices landed
    }
  }

  // epilogue: demod scale, scatter to NHWC (D layout: VGPR r = row m, lanes = cols)
  const int bb = (tile * 16) / HW;  // whole tile is one sample (HW % 16 == 0)
#pragma unroll
  for (int f = 0; f < NF; ++f) {
    int nn = n0 + f * 16 + m;
    float dm = demod[bb * Cout + nn];
#pragma unroll
    for (int r = 0; r < 8; ++r) {
      int pp = tile * 16 + half * 8 + r;
      yraw[(size_t)pp * Cout + nn] = acc[f][r] * dm;
    }
  }
}

// ---------------- host orchestration ----------------

struct LC { int cin, cout, k, up, dem; };
static const LC g_cfgs[20] = {
  {512, 512, 3, 0, 1}, {512, 3, 1, 0, 0},
  {512, 512, 3, 1, 1}, {512, 512, 3, 0, 1}, {512, 3, 1, 0, 0},
  {512, 256, 3, 1, 1}, {256, 256, 3, 0, 1}, {256, 3, 1, 0, 0},
  {256, 128, 3, 1, 1}, {128, 128, 3, 0, 1}, {128, 3, 1, 0, 0},
  {128,  64, 3, 1, 1}, { 64,  64, 3, 0, 1}, { 64, 3, 1, 0, 0},
  { 64,  32, 3, 1, 1}, { 32,  32, 3, 0, 1}, { 32, 3, 1, 0, 0},
  { 32,  16, 3, 1, 1}, { 16,  16, 3, 0, 1}, { 16, 3, 1, 0, 0},
};

static inline int ceil_div(int a, int b) { return (a + b - 1) / b; }

extern "C" void kernel_launch(void* const* d_in, const int* in_sizes, int n_in,
                              void* d_out, int out_size, void* d_ws, size_t ws_size,
                              hipStream_t stream) {
  (void)in_sizes; (void)n_in; (void)out_size; (void)ws_size;
  const float* z = (const float*)d_in[0];
  const float* learned = (const float*)d_in[17];
  auto mapW = [&](int i) { return (const float*)d_in[1 + 2 * i]; };
  auto mapB = [&](int i) { return (const float*)d_in[2 + 2 * i]; };
  auto convW = [&](int l) { return (const float*)d_in[18 + 5 * l + 0]; };
  auto convB = [&](int l) { return (const float*)d_in[18 + 5 * l + 1]; };
  auto styW  = [&](int l) { return (const float*)d_in[18 + 5 * l + 2]; };
  auto styB  = [&](int l) { return (const float*)d_in[18 + 5 * l + 3]; };
  auto noiW  = [&](int l) { return (const float*)d_in[18 + 5 * l + 4]; };

  // workspace carve-up
  char* base = (char*)d_ws;
  size_t off = 0;
  auto take = [&](size_t bytes) {
    char* p = base + off;
    off += (bytes + 255) & ~(size_t)255;
    return p;
  };
  __bf16* MAPW = (__bf16*)take(8ull * 512 * 512 * 2);
  __bf16* ACT0 = (__bf16*)take(16 * 512 * 2);
  __bf16* ACT1 = (__bf16*)take(16 * 512 * 2);
  float*  WLAT = (float*)take(16 * 512 * 4);
  float*  STYLE = (float*)take(16 * 512 * 4);
  float*  DEMOD = (float*)take(16 * 512 * 4);
  float*  WSQ  = (float*)take(512ull * 512 * 4);
  __bf16* WPK  = (__bf16*)take(9ull * 512 * 512 * 2);
  float*  X0   = (float*)take(67108864ull);
  float*  X1   = (float*)take(67108864ull);
  __bf16* XS   = (__bf16*)take(67108864ull);
  float*  YRAW = (float*)take(67108864ull);
  float*  YRW2 = (float*)take(67108864ull);
  float*  RBUF = (float*)take(12582912ull);
  float*  YA   = (float*)take(12582912ull);
  float*  YB   = (float*)take(12582912ull);

  const int T = 256;
  auto g1 = [&](int n) { return dim3(ceil_div(n, T)); };

  // 1) pack mapping weights + z to bf16
  for (int i = 0; i < 8; ++i)
    k_pack_bf16<<<g1(512 * 512), T, 0, stream>>>(mapW(i), MAPW + (size_t)i * 512 * 512, 512 * 512);
  k_pack_bf16<<<g1(16 * 512), T, 0, stream>>>(z, ACT0, 16 * 512);

  // 2) mapping MLP (8 WMMA GEMM layers), final layer also writes f32 latent
  __bf16* ain = ACT0; __bf16* aout = ACT1;
  for (int i = 0; i < 8; ++i) {
    k_map_layer<<<dim3(8), 32, 0, stream>>>(ain, MAPW + (size_t)i * 512 * 512, mapB(i),
                                            aout, WLAT, i == 7 ? 1 : 0);
    __bf16* t = ain; ain = aout; aout = t;
  }

  // 3) synthesis
  k_init<<<g1(16 * 4 * 4 * 512), T, 0, stream>>>(learned, X0);
  float* X = X0; float* Xalt = X1;
  float* Ycur = YA; float* Yalt = YB;
  int H = 4, Cx = 512, have_y = 0;

  for (int l = 0; l < 20; ++l) {
    const LC c = g_cfgs[l];
    if (c.dem) {  // styled 3x3 conv block
      int Cpad = c.cin < 32 ? 32 : c.cin;
      int Hout = c.up ? H * 2 : H;
      k_style<<<g1(16 * Cpad), T, 0, stream>>>(WLAT, styW(l), styB(l), STYLE, c.cin, Cpad);
      k_prepack<<<g1(c.cout * Cpad), T, 0, stream>>>(convW(l), WPK, WSQ, c.cout, c.cin, Cpad);
      k_demod<<<g1(16 * c.cout), T, 0, stream>>>(STYLE, WSQ, DEMOD, c.cout, Cpad);
      k_modulate<<<g1(16 * Hout * Hout * Cpad), T, 0, stream>>>(X, STYLE, XS, H, Cx,
                                                                Hout, Cpad, c.up);
      int nf = c.cout / 16; if (nf > 4) nf = 4;
      dim3 grid(Hout * Hout / 8, c.cout / (nf * 16));  // ntiles = H*H (mult of 8)
      if (nf == 4)
        k_conv3<4><<<grid, 256, 0, stream>>>(XS, WPK, DEMOD, YRAW, Hout, Cpad, c.cout);
      else if (nf == 2)
        k_conv3<2><<<grid, 256, 0, stream>>>(XS, WPK, DEMOD, YRAW, Hout, Cpad, c.cout);
      else
        k_conv3<1><<<grid, 256, 0, stream>>>(XS, WPK, DEMOD, YRAW, Hout, Cpad, c.cout);
      const float* ysrc = YRAW;
      if (c.up) {
        k_blur<<<g1(16 * Hout * Hout * c.cout), T, 0, stream>>>(YRAW, nullptr, YRW2,
                                                                Hout, c.cout, 0);
        ysrc = YRW2;
      }
      k_finish<<<g1(16 * Hout * Hout * c.cout), T, 0, stream>>>(ysrc, convB(l), noiW(l),
                                                                Xalt, Hout, c.cout, l);
      float* t = X; X = Xalt; Xalt = t;
      H = Hout; Cx = c.cout;
    } else {  // to-RGB + skip accumulate
      k_style<<<g1(16 * c.cin), T, 0, stream>>>(WLAT, styW(l), styB(l), STYLE, c.cin, c.cin);
      k_rgb<<<g1(16 * H * H), T, 0, stream>>>(X, STYLE, convW(l), convB(l), noiW(l),
                                              RBUF, H, c.cin, l);
      if (!have_y) {
        k_copyf<<<g1(16 * H * H * 3), T, 0, stream>>>(RBUF, Ycur, 16 * H * H * 3);
        have_y = 1;
      } else {
        k_blur<<<g1(16 * H * H * 3), T, 0, stream>>>(Ycur, RBUF, Yalt, H, 3, 1);
        float* t = Ycur; Ycur = Yalt; Yalt = t;
      }
    }
  }

  k_output<<<g1(16 * 3 * 256 * 256), T, 0, stream>>>(Ycur, (float*)d_out);
}
